// AutoCorrelation_84164179133273
// MI455X (gfx1250) — matrix operand order repaired and checked by reading, not compiled, not run on hardware
//
#include <hip/hip_runtime.h>
#include <hip/hip_bf16.h>

typedef _Float16 v16h __attribute__((ext_vector_type(16)));
typedef _Float16 v8h  __attribute__((ext_vector_type(8)));
typedef float    v8f  __attribute__((ext_vector_type(8)));

#define B_    8
#define L_    4096
#define H_    8
#define D_    32
#define LMASK 4095
#define TOPK  8

// ---------------------------------------------------------------------------
// Kernel 1: convert fp32 [B,L,H,D] -> f16 [B,H,L,D] for Q and K (coalesced on d)
// ---------------------------------------------------------------------------
__global__ void cvt_kernel(const float* __restrict__ q, const float* __restrict__ k,
                           _Float16* __restrict__ qh, _Float16* __restrict__ kh) {
    int e = blockIdx.x * blockDim.x + threadIdx.x;   // over B*H*L*D = 2^23
    int d = e & 31;
    int l = (e >> 5) & 4095;
    int h = (e >> 17) & 7;
    int b = e >> 20;
    int src = ((b * L_ + l) * H_ + h) * D_ + d;
    qh[e] = (_Float16)q[src];
    kh[e] = (_Float16)k[src];
}

// ---------------------------------------------------------------------------
// Kernel 2: corr[bh, tau] = sum_{l,d} q[(l+tau)%L, d] * k[l, d]
// One wave per (bh, delta) diagonal band; delta = 16 * band.
// C[i,j] of the 16x16x32 WMMA tile always maps to the same
// tau = delta + (r + 8*hi) - (lane&15), so the WMMA C operand accumulates
// per-tau partial sums across all 256 c0 steps; scatter once with f32 atomics.
// Constant trip count (256) keeps the loop wave-uniform so the backend
// unrolls x8 and software-pipelines the global_load_b128 clauses; bh/dlt are
// scalarized via readfirstlane so base addresses live in SGPRs.
// ---------------------------------------------------------------------------
__global__ void corr_kernel(const _Float16* __restrict__ qh,
                            const _Float16* __restrict__ kh,
                            float* __restrict__ corr) {
    int wave = (blockIdx.x * blockDim.x + threadIdx.x) >> 5;  // 0..16383
    int lane = threadIdx.x & 31;
    int bh   = __builtin_amdgcn_readfirstlane(wave >> 8);          // 0..63
    int dlt  = __builtin_amdgcn_readfirstlane((wave & 255) << 4);  // diag offset
    int m    = lane & 15;          // A row (M) / B col (N) within tile
    int hi   = lane >> 4;          // half-wave selects K-subrange

    const _Float16* qb = qh + (size_t)bh * (L_ * D_);
    const _Float16* kb = kh + (size_t)bh * (L_ * D_);

    v8f acc = {0.f, 0.f, 0.f, 0.f, 0.f, 0.f, 0.f, 0.f};

    const int rowBase = dlt + m;           // A row before wrap
    const unsigned aSub = 16u * hi;        // half-offset within A row
    const unsigned bOffBase = (unsigned)m * D_ + 16u * hi;

    for (int c0 = 0; c0 < L_; c0 += 16) {
        // A fragment: q row (c0+dlt+m) mod L; elems 0..15 = K[16hi..16hi+15]
        unsigned arow = (unsigned)(c0 + rowBase) & LMASK;
        unsigned aoff = arow * D_ + aSub;          // in halves, 32-bit offset
        v8h alo = *(const v8h*)(qb + aoff);
        v8h ahi = *(const v8h*)(qb + aoff + 8);
        // B fragment: k row (c0+m), column lane; elems 0..15 = K[16hi..16hi+15]
        unsigned boff = bOffBase + (unsigned)c0 * D_;
        v8h blo = *(const v8h*)(kb + boff);
        v8h bhi = *(const v8h*)(kb + boff + 8);

        v16h a  = __builtin_shufflevector(alo, ahi, 0,1,2,3,4,5,6,7,8,9,10,11,12,13,14,15);
        v16h bm = __builtin_shufflevector(blo, bhi, 0,1,2,3,4,5,6,7,8,9,10,11,12,13,14,15);

        acc = __builtin_amdgcn_wmma_f32_16x16x32_f16(
                  /*neg_a=*/false, a, /*neg_b=*/false, bm,
                  /*c_mod=*/(short)0, acc, /*reuse_a=*/false, /*reuse_b=*/false);
    }

    float* crow = corr + bh * L_;
    int baseTau = dlt + 8 * hi - m;   // tau for acc[r] is baseTau + r (mod L)
#pragma unroll
    for (int r = 0; r < 8; ++r) {
        int tau = (baseTau + r) & LMASK;
        __hip_atomic_fetch_add(&crow[tau], acc[r],
                               __ATOMIC_RELAXED, __HIP_MEMORY_SCOPE_AGENT);
    }
}

// ---------------------------------------------------------------------------
// Kernel 3: per (b,h): top-8 of corr over tau, softmax, then
// out[b,l,h,d] = sum_k w_k * v[b,(l+delay_k)%L,h,d]. One block per (b,h).
// ---------------------------------------------------------------------------
__global__ void topk_agg_kernel(const float* __restrict__ corr,
                                const float* __restrict__ v,
                                float* __restrict__ out) {
    int bh = blockIdx.x;
    int b = bh >> 3, h = bh & 7;
    int tid = threadIdx.x;  // 256 threads

    __shared__ float sc[L_];
    __shared__ float rv[256];
    __shared__ int   ri[256];
    __shared__ float wk[TOPK];
    __shared__ int   dk[TOPK];

    const float* crow = corr + bh * L_;
    for (int i = tid; i < L_; i += 256) sc[i] = crow[i];
    __syncthreads();

    for (int k = 0; k < TOPK; ++k) {
        float bv = -3.4e38f; int bi = 0;
        for (int i = tid; i < L_; i += 256) {
            float x = sc[i];
            if (x > bv) { bv = x; bi = i; }
        }
        rv[tid] = bv; ri[tid] = bi;
        __syncthreads();
        for (int s = 128; s > 0; s >>= 1) {
            if (tid < s && rv[tid + s] > rv[tid]) {
                rv[tid] = rv[tid + s]; ri[tid] = ri[tid + s];
            }
            __syncthreads();
        }
        if (tid == 0) { wk[k] = rv[0]; dk[k] = ri[0]; sc[ri[0]] = -3.4e38f; }
        __syncthreads();
    }

    if (tid == 0) {
        float mx = wk[0];
        for (int k = 1; k < TOPK; ++k) mx = fmaxf(mx, wk[k]);
        float s = 0.f, e[TOPK];
        for (int k = 0; k < TOPK; ++k) { e[k] = expf(wk[k] - mx); s += e[k]; }
        for (int k = 0; k < TOPK; ++k) wk[k] = e[k] / s;
    }
    __syncthreads();

    // Weighted circular-gather aggregation, float4 vectorized (D=32 -> 8 chunks)
    const float4* v4 = (const float4*)v;
    float4*       o4 = (float4*)out;
    for (int e2 = tid; e2 < L_ * 8; e2 += 256) {
        int l = e2 >> 3;
        int c = e2 & 7;
        float sx = 0.f, sy = 0.f, sz = 0.f, sw = 0.f;
#pragma unroll
        for (int k = 0; k < TOPK; ++k) {
            int lk = (l + dk[k]) & LMASK;
            float4 x = v4[((size_t)(b * L_ + lk) * H_ + h) * 8 + c];
            float w = wk[k];
            sx += w * x.x; sy += w * x.y; sz += w * x.z; sw += w * x.w;
        }
        float4 r; r.x = sx; r.y = sy; r.z = sz; r.w = sw;
        o4[((size_t)(b * L_ + l) * H_ + h) * 8 + c] = r;
    }
}

// ---------------------------------------------------------------------------
extern "C" void kernel_launch(void* const* d_in, const int* in_sizes, int n_in,
                              void* d_out, int out_size, void* d_ws, size_t ws_size,
                              hipStream_t stream) {
    (void)in_sizes; (void)n_in; (void)out_size; (void)ws_size;
    const float* q = (const float*)d_in[0];
    const float* k = (const float*)d_in[1];
    const float* v = (const float*)d_in[2];
    float* out = (float*)d_out;

    char* ws = (char*)d_ws;
    const size_t BHLD = (size_t)B_ * H_ * L_ * D_;       // 8,388,608
    _Float16* qh  = (_Float16*)ws;                       // 16 MB
    _Float16* kh  = (_Float16*)(ws + BHLD * 2);          // 16 MB
    float*    cor = (float*)(ws + BHLD * 4);             // 1 MB

    // 1) fp32 -> f16 transpose for Q,K
    cvt_kernel<<<(int)(BHLD / 256), 256, 0, stream>>>(q, k, qh, kh);
    // 2) zero corr accumulator
    hipMemsetAsync(cor, 0, (size_t)B_ * H_ * L_ * sizeof(float), stream);
    // 3) WMMA diagonal-band correlation: 16384 waves, 8 waves/block
    corr_kernel<<<2048, 256, 0, stream>>>(qh, kh, cor);
    // 4) top-8 + softmax + weighted circular gather
    topk_agg_kernel<<<B_ * H_, 256, 0, stream>>>(cor, v, out);
}
